// LinearAttentionLayer_49005576847619
// MI455X (gfx1250) — matrix operand order repaired and checked by reading, not compiled, not run on hardware
//
#include <hip/hip_runtime.h>
#include <hip/hip_bf16.h>

#define BB 4
#define NN 4096
#define CC 1024
#define HH 16
#define DD 64
#define MM (BB * NN)
#define BH (BB * HH)
#define EPSV 1e-6f

// LDS W-tile row pitch: 32 data elems + 8 pad elems (80B rows -> banks 20n%64
// distinct for n=0..15 => conflict-free fragment reads, 16B aligned chunks)
#define WPITCH 40

typedef __attribute__((ext_vector_type(8)))  __bf16 v8bf;
typedef __attribute__((ext_vector_type(16))) __bf16 v16bf;
typedef __attribute__((ext_vector_type(8)))  float  v8f;

// ---- WMMA fragment loaders (CDNA5 16x16x32 bf16 layouts, wave32) ----
// A (16x32, MxK): lane = m + 16*half; lane holds K = half*16..+15
__device__ __forceinline__ v16bf load_a16x32(const __bf16* base, int stride,
                                             int row, int k0, int lane) {
  const int m = lane & 15, half = lane >> 4;
  const __bf16* p = base + (size_t)(row + m) * stride + (k0 + half * 16);
  v8bf lo = *(const v8bf*)(p);
  v8bf hi = *(const v8bf*)(p + 8);
  return __builtin_shufflevector(lo, hi, 0,1,2,3,4,5,6,7,8,9,10,11,12,13,14,15);
}

// B (32x16, KxN) from a row-major [Nout x K] matrix: lane = n + 16*half holds
// K = half*16..+15 of column n (= row n of the matrix, contiguous).
__device__ __forceinline__ v16bf load_b32x16(const __bf16* base, int stride,
                                             int col, int k0, int lane) {
  const int n = lane & 15, half = lane >> 4;
  const __bf16* p = base + (size_t)(col + n) * stride + (k0 + half * 16);
  return *(const v16bf*)(p);
}

// B fragment from the LDS-staged W tile (row-major [64][WPITCH], 32 valid k)
__device__ __forceinline__ v16bf lds_b_frag(const __bf16* s, int j, int lane) {
  const int n = j * 16 + (lane & 15), half = lane >> 4;
  const __bf16* p = s + n * WPITCH + half * 16;
  v8bf lo = *(const v8bf*)(p);
  v8bf hi = *(const v8bf*)(p + 8);
  return __builtin_shufflevector(lo, hi, 0,1,2,3,4,5,6,7,8,9,10,11,12,13,14,15);
}

__device__ __forceinline__ float phi_act(float x) {
  // elu(x)+1 : x>0 ? x+1 : exp(x)
  return x > 0.0f ? x + 1.0f : __expf(x);
}

// ---- CDNA5 async Global->LDS copy of one 64x32 bf16 W tile (4KB) ----
// 256 threads, each moves one 16B chunk; tracked by ASYNCcnt.
__device__ __forceinline__ void async_copy_w_tile(const __bf16* W, int colBase,
                                                  int k0, __bf16* sdst, int tid) {
  const int n = tid >> 2, c = tid & 3;
  const __bf16* g = W + (size_t)(colBase + n) * CC + (k0 + c * 8);
  unsigned lds = (unsigned)(uintptr_t)(sdst + n * WPITCH + c * 8);
  asm volatile("global_load_async_to_lds_b128 %0, %1, off"
               :: "v"(lds), "v"(g) : "memory");
}

__device__ __forceinline__ void wait_async0() {
  asm volatile("s_wait_asynccnt 0x0" ::: "memory");
}

// ---- f32 -> bf16 conversion ----
__global__ void cvt_f32_bf16(const float* __restrict__ in,
                             __bf16* __restrict__ out, int n) {
  int i = blockIdx.x * blockDim.x + threadIdx.x;
  if (i < n) out[i] = (__bf16)in[i];
}

// ---- Tiled bf16 GEMM: out = act(X[M x CC] @ W^T + b), W row-major [CC x CC].
// 256 threads = 8 waves; block tile 256 rows x 64 cols; wave tile 32x64
// (8 accumulators). W tile double-buffered in LDS via async Global->LDS.
// mode 0: f32 store [M x CC] | mode 1: bf16 [B,H,N,D] | mode 2: bf16 [B,H,D,N]
__global__ __launch_bounds__(256)
void gemm_proj_kernel(const __bf16* __restrict__ X, const __bf16* __restrict__ W,
                      const float* __restrict__ bias, int act, int mode,
                      float* __restrict__ outF, __bf16* __restrict__ outB) {
  __shared__ __attribute__((aligned(16))) __bf16 sW[2][64 * WPITCH];

  const int tid  = threadIdx.x;
  const int lane = tid & 31;
  const int wave = tid >> 5;
  const int rowBase = blockIdx.x * 256 + wave * 32;
  const int colBase = blockIdx.y * 64;

  v8f acc[2][4] = {};

  async_copy_w_tile(W, colBase, 0, &sW[0][0], tid);

  for (int k0 = 0; k0 < CC; k0 += 32) {
    const int cur = (k0 >> 5) & 1;
    wait_async0();        // my chunk of the current tile has landed
    __syncthreads();      // everyone's chunks landed; prev compute done reading cur^1
    if (k0 + 32 < CC)
      async_copy_w_tile(W, colBase, k0 + 32, &sW[cur ^ 1][0], tid);

    // A tiles stream from global; all four B fragments preloaded from LDS so
    // the ds loads clause under one wait and the 8 WMMAs issue back-to-back.
    v16bf a0 = load_a16x32(X, CC, rowBase, k0, lane);
    v16bf a1 = load_a16x32(X, CC, rowBase + 16, k0, lane);
    v16bf b[4];
#pragma unroll
    for (int j = 0; j < 4; ++j) b[j] = lds_b_frag(&sW[cur][0], j, lane);
#pragma unroll
    for (int j = 0; j < 4; ++j) {
      acc[0][j] = __builtin_amdgcn_wmma_f32_16x16x32_bf16(false, a0, false, b[j], (short)0, acc[0][j], false, false);
      acc[1][j] = __builtin_amdgcn_wmma_f32_16x16x32_bf16(false, a1, false, b[j], (short)0, acc[1][j], false, false);
    }
  }

  const int half = lane >> 4, nl = lane & 15;
#pragma unroll
  for (int s = 0; s < 2; ++s) {
#pragma unroll
    for (int j = 0; j < 4; ++j) {
#pragma unroll
      for (int r = 0; r < 8; ++r) {
        const int row = rowBase + s * 16 + r + 8 * half;  // C/D: m = r + 8*half
        const int col = colBase + j * 16 + nl;            // n = lane & 15
        float v = acc[s][j][r] + bias[col];
        if (act) v = phi_act(v);
        if (mode == 0) {
          outF[(size_t)row * CC + col] = v;
        } else {
          const int b2 = row / NN, n = row % NN;
          const int h = col / DD, d = col % DD;
          if (mode == 1)
            outB[(((size_t)(b2 * HH + h)) * NN + n) * DD + d] = (__bf16)v;
          else
            outB[(((size_t)(b2 * HH + h)) * DD + d) * NN + n] = (__bf16)v;
        }
      }
    }
  }
}

// ---- kv[b,h] = K^T V : 64x64 per head, contraction over N=4096.
// 64 blocks (one per b,h), 16 waves; wave owns one 16x16 tile, 128 WMMAs.
__global__ __launch_bounds__(512)
void kv_kernel(const __bf16* __restrict__ Kt, const __bf16* __restrict__ Vt,
               __bf16* __restrict__ kvt) {
  const int bh = blockIdx.x;
  const int lane = threadIdx.x & 31;
  const int wave = threadIdx.x >> 5;
  const int dt = (wave >> 2) * 16;
  const int et = (wave & 3) * 16;
  const __bf16* Kb = Kt + (size_t)bh * DD * NN;
  const __bf16* Vb = Vt + (size_t)bh * DD * NN;

  v8f acc = {};
  for (int n0 = 0; n0 < NN; n0 += 32) {
    v16bf a = load_a16x32(Kb, NN, dt, n0, lane);
    v16bf b = load_b32x16(Vb, NN, et, n0, lane);
    acc = __builtin_amdgcn_wmma_f32_16x16x32_bf16(false, a, false, b, (short)0, acc, false, false);
  }
  const int half = lane >> 4, nl = lane & 15;
#pragma unroll
  for (int r = 0; r < 8; ++r) {
    const int d = dt + r + 8 * half;
    const int e = et + nl;
    kvt[((size_t)bh * DD + e) * DD + d] = (__bf16)acc[r];   // transposed store
  }
}

// ---- ksum[b,h,d] = sum_n K[n,d]
__global__ __launch_bounds__(64)
void ksum_kernel(const __bf16* __restrict__ Kt, float* __restrict__ ksum) {
  const int bh = blockIdx.x;
  const int d = threadIdx.x;
  const __bf16* p = Kt + ((size_t)bh * DD + d) * NN;
  float s = 0.f;
  for (int n = 0; n < NN; n += 8) {
    v8bf v = *(const v8bf*)(p + n);
#pragma unroll
    for (int i = 0; i < 8; ++i) s += (float)v[i];
  }
  ksum[bh * DD + d] = s;
}

// ---- z[b,h,n] = 1 / (q . ksum + eps)
__global__ __launch_bounds__(256)
void z_kernel(const __bf16* __restrict__ Q, const float* __restrict__ ksum,
              float* __restrict__ z) {
  const int idx = blockIdx.x * 256 + threadIdx.x;
  const int bh = idx / NN, n = idx % NN;
  const __bf16* q = Q + ((size_t)bh * NN + n) * DD;
  const float* ks = ksum + bh * DD;
  float s = 0.f;
#pragma unroll
  for (int d = 0; d < DD; ++d) s += (float)q[d] * ks[d];
  z[idx] = 1.0f / (s + EPSV);
}

// ---- attn = (Q @ kv) * z -> Y[B,N,C] bf16 (heads re-interleaved)
__global__ __launch_bounds__(256)
void attn_kernel(const __bf16* __restrict__ Q, const __bf16* __restrict__ kvt,
                 const float* __restrict__ z, __bf16* __restrict__ Y) {
  const int bh = blockIdx.x;
  const int lane = threadIdx.x & 31;
  const int wave = threadIdx.x >> 5;
  const int n0 = blockIdx.y * 128 + wave * 16;
  const __bf16* Qb = Q + (size_t)bh * NN * DD;
  const __bf16* kb = kvt + (size_t)bh * DD * DD;

  v8f acc0 = {}, acc1 = {}, acc2 = {}, acc3 = {};
  for (int d0 = 0; d0 < DD; d0 += 32) {
    v16bf a  = load_a16x32(Qb, DD, n0, d0, lane);
    v16bf b0 = load_b32x16(kb, DD,  0, d0, lane);
    v16bf b1 = load_b32x16(kb, DD, 16, d0, lane);
    v16bf b2 = load_b32x16(kb, DD, 32, d0, lane);
    v16bf b3 = load_b32x16(kb, DD, 48, d0, lane);
    acc0 = __builtin_amdgcn_wmma_f32_16x16x32_bf16(false, a, false, b0, (short)0, acc0, false, false);
    acc1 = __builtin_amdgcn_wmma_f32_16x16x32_bf16(false, a, false, b1, (short)0, acc1, false, false);
    acc2 = __builtin_amdgcn_wmma_f32_16x16x32_bf16(false, a, false, b2, (short)0, acc2, false, false);
    acc3 = __builtin_amdgcn_wmma_f32_16x16x32_bf16(false, a, false, b3, (short)0, acc3, false, false);
  }

  const int half = lane >> 4, nl = lane & 15;
  const int b = bh / HH, h = bh % HH;
  v8f accs[4] = {acc0, acc1, acc2, acc3};
#pragma unroll
  for (int j = 0; j < 4; ++j) {
#pragma unroll
    for (int r = 0; r < 8; ++r) {
      const int n = n0 + r + 8 * half;
      const int e = j * 16 + nl;
      const float v = accs[j][r] * z[bh * NN + n];
      Y[((size_t)(b * NN + n)) * CC + h * DD + e] = (__bf16)v;
    }
  }
}

extern "C" void kernel_launch(void* const* d_in, const int* in_sizes, int n_in,
                              void* d_out, int out_size, void* d_ws, size_t ws_size,
                              hipStream_t stream) {
  const float* x  = (const float*)d_in[0];
  const float* Wq = (const float*)d_in[1];
  const float* bq = (const float*)d_in[2];
  const float* Wk = (const float*)d_in[3];
  const float* bk = (const float*)d_in[4];
  const float* Wv = (const float*)d_in[5];
  const float* bv = (const float*)d_in[6];
  const float* Wo = (const float*)d_in[7];
  const float* bo = (const float*)d_in[8];
  float* out = (float*)d_out;

  char* ws = (char*)d_ws;
  size_t off = 0;
  auto alloc = [&](size_t bytes) {
    char* p = ws + off;
    off += (bytes + 255) & ~(size_t)255;
    return p;
  };
  __bf16* xb   = (__bf16*)alloc((size_t)MM * CC * 2);   // x bf16 (reused as Y)
  __bf16* wqb  = (__bf16*)alloc((size_t)CC * CC * 2);
  __bf16* wkb  = (__bf16*)alloc((size_t)CC * CC * 2);
  __bf16* wvb  = (__bf16*)alloc((size_t)CC * CC * 2);
  __bf16* wob  = (__bf16*)alloc((size_t)CC * CC * 2);
  __bf16* Qb   = (__bf16*)alloc((size_t)BH * NN * DD * 2);  // [B,H,N,D]
  __bf16* Ktb  = (__bf16*)alloc((size_t)BH * DD * NN * 2);  // [B,H,D,N]
  __bf16* Vtb  = (__bf16*)alloc((size_t)BH * DD * NN * 2);  // [B,H,D,N]
  __bf16* kvtb = (__bf16*)alloc((size_t)BH * DD * DD * 2);  // [B,H,E,D]
  float*  ksum = (float*)alloc((size_t)BH * DD * 4);
  float*  zbuf = (float*)alloc((size_t)BH * NN * 4);
  __bf16* Yb   = xb;   // alias: x bf16 dead after V projection (stream-ordered)

  // 1) conversions
  cvt_f32_bf16<<<(MM * CC) / 256, 256, 0, stream>>>(x,  xb,  MM * CC);
  cvt_f32_bf16<<<(CC * CC) / 256, 256, 0, stream>>>(Wq, wqb, CC * CC);
  cvt_f32_bf16<<<(CC * CC) / 256, 256, 0, stream>>>(Wk, wkb, CC * CC);
  cvt_f32_bf16<<<(CC * CC) / 256, 256, 0, stream>>>(Wv, wvb, CC * CC);
  cvt_f32_bf16<<<(CC * CC) / 256, 256, 0, stream>>>(Wo, wob, CC * CC);

  // 2) projections (WMMA GEMMs, async-LDS staged weights, fused bias + phi)
  dim3 ggrid(MM / 256, CC / 64);
  gemm_proj_kernel<<<ggrid, 256, 0, stream>>>(xb, wqb, bq, 1, 1, nullptr, Qb);
  gemm_proj_kernel<<<ggrid, 256, 0, stream>>>(xb, wkb, bk, 1, 2, nullptr, Ktb);
  gemm_proj_kernel<<<ggrid, 256, 0, stream>>>(xb, wvb, bv, 0, 2, nullptr, Vtb);

  // 3) kv = K^T V per head (WMMA), 4) ksum, 5) z
  kv_kernel<<<BH, 512, 0, stream>>>(Ktb, Vtb, kvtb);
  ksum_kernel<<<BH, 64, 0, stream>>>(Ktb, ksum);
  z_kernel<<<(BH * NN) / 256, 256, 0, stream>>>(Qb, ksum, zbuf);

  // 6) attn = (Q @ kv) * z  (WMMA)
  attn_kernel<<<dim3(BH, NN / 128), 256, 0, stream>>>(Qb, kvtb, zbuf, Yb);

  // 7) out = Y @ Wo^T + bo  (WMMA, f32 output)
  gemm_proj_kernel<<<ggrid, 256, 0, stream>>>(Yb, wob, bo, 0, 0, out, nullptr);
}
